// Model_9955734192204
// MI455X (gfx1250) — matrix operand order repaired and checked by reading, not compiled, not run on hardware
//
#include <hip/hip_runtime.h>
#include <hip/hip_bf16.h>
#include <math.h>

typedef __attribute__((ext_vector_type(16))) _Float16 v16h;
typedef __attribute__((ext_vector_type(8)))  _Float16 h8;
typedef __attribute__((ext_vector_type(8)))  float    v8f;

namespace {

constexpr int Bsz    = 512;
constexpr int HC     = 20;   // 28-9+1
constexpr int HP     = 6;    // (20-9)/2+1
constexpr int C1     = 256;  // conv1 out channels
constexpr int KK     = 81;   // 9*9 taps
constexpr int KPAD   = 96;   // 81 padded to multiple of 32
constexpr int CPRIM  = 256;  // NPRIM*PATOMS
constexpr int NPRIM  = 32;
constexpr int PAT    = 8;
constexpr int NCLASS = 10;
constexpr int DAT    = 16;
constexpr int D0     = 160;  // NCLASS*DAT
constexpr int NCAPS  = 1152; // 32*36
constexpr int HPIX   = HC * HC;       // 400
constexpr int PPIX   = HP * HP;       // 36
constexpr int N1     = Bsz * HPIX;    // 204800 (conv1 GEMM N)
constexpr int N2     = Bsz * PPIX;    // 18432  (prim GEMM N)
constexpr int NPIXI  = 28 * 28;       // 784

// ---------------- WMMA helpers (ISA 7.12.2 layouts, wave32) ----------------
// A frag (16-bit 16x32): lane -> m = lane&15 ; element j -> k = 16*hi + j (this file's order)
//   => two contiguous 8-half chunks at k = 2*koffA and k = 2*koffA+8
// B frag (16-bit 32x16): lane -> n = lane&15 ; element j -> k = 16*hi + j (this file's order, same as A)
//   => one contiguous 16-half chunk at k = koffB
// C/D: reg r -> m = r + 8*hi ; n = lane&15
__device__ __forceinline__ v8f wmma_f16(v16h a, v16h b, v8f c) {
  return __builtin_amdgcn_wmma_f32_16x16x32_f16(false, a, false, b, (short)0, c,
                                                false, false);
}
__device__ __forceinline__ v8f zero8() {
  v8f z;
#pragma unroll
  for (int i = 0; i < 8; ++i) z[i] = 0.0f;
  return z;
}
__device__ __forceinline__ v16h zero16h() {
  v16h z;
#pragma unroll
  for (int i = 0; i < 16; ++i) z[i] = (_Float16)0.0f;
  return z;
}
__device__ __forceinline__ v16h concat8(h8 lo, h8 hi) {
  return __builtin_shufflevector(lo, hi, 0, 1, 2, 3, 4, 5, 6, 7, 8, 9, 10, 11,
                                 12, 13, 14, 15);
}
// A fragment: two contiguous b128 loads
__device__ __forceinline__ v16h load_afrag(const _Float16* row, int k, int koffA) {
  return concat8(*(const h8*)(row + k + 2 * koffA), *(const h8*)(row + k + 8 + 2 * koffA));
}

// ---------------- small utility kernels ----------------
__global__ void zero_f32_kernel(float* __restrict__ p, int n) {
  int i = blockIdx.x * blockDim.x + threadIdx.x;
  if (i < n) p[i] = 0.0f;
}
// conv1_w (256,1,9,9) -> f16 [256][96] zero padded
__global__ void pack_w1_kernel(const float* __restrict__ w, _Float16* __restrict__ d) {
  int i = blockIdx.x * blockDim.x + threadIdx.x;
  if (i >= C1 * KPAD) return;
  int m = i / KPAD, k = i - m * KPAD;
  d[i] = (_Float16)(k < KK ? w[m * KK + k] : 0.0f);
}
// im2col for conv1, fused f32->f16: xcol[n][k], n = b*400 + oy*20 + ox, k in [0,96)
__global__ void im2col1_kernel(const float* __restrict__ x, _Float16* __restrict__ xcol) {
  int i = blockIdx.x * blockDim.x + threadIdx.x;
  if (i >= N1 * KPAD) return;
  int n = i / KPAD, k = i - n * KPAD;
  int b = n / HPIX, s = n - b * HPIX;
  int oy = s / HC, ox = s - oy * HC;
  float v = 0.0f;
  if (k < KK) {
    int ky = k / 9, kx = k - ky * 9;
    v = x[b * NPIXI + (oy + ky) * 28 + (ox + kx)];
  }
  xcol[i] = (_Float16)v;
}
// prim_w (256,256,9,9)[m][c][t] -> f16 [81][256][256] (tap-major, row-major per tap)
__global__ void pack_pw_kernel(const float* __restrict__ w, _Float16* __restrict__ d) {
  int i = blockIdx.x * blockDim.x + threadIdx.x;
  if (i >= KK * CPRIM * C1) return;
  int t = i >> 16;  // /65536
  int r = i & 65535;
  int m = r >> 8, c = r & 255;
  d[i] = (_Float16)w[(m * C1 + c) * KK + t];
}
// transpose f32 [K][N] -> f16 [N][K] (decoder weights, K-contiguous B frags)
__global__ void tr_f16_kernel(const float* __restrict__ s, _Float16* __restrict__ d,
                              int K, int N) {
  int i = blockIdx.x * blockDim.x + threadIdx.x;
  if (i >= K * N) return;
  int n = i / K, k = i - n * K;
  d[i] = (_Float16)s[k * N + n];
}

// ---------------- conv1 GEMM: C[256 x 204800], A=[256][96], B=xcol[n][96] ----
// epilogue: relu(acc+bias) -> NHWC f16 Hout[(b*400+s)*256 + c], 8 channels/store
__global__ __launch_bounds__(256) void conv1_gemm_kernel(
    const _Float16* __restrict__ W1, const _Float16* __restrict__ Xc,
    const float* __restrict__ bias, _Float16* __restrict__ Hout) {
  const int lane = threadIdx.x & 31;
  const int wv = threadIdx.x >> 5;
  const int wm = wv >> 2, wn = wv & 3;
  const int m0 = blockIdx.x * 64 + wm * 32;
  const int n0 = blockIdx.y * 256 + wn * 64;
  const int lm = lane & 15;
  const int hi = (lane >> 4) & 1;
  const int koffA = hi * 8, koffB = hi * 16;

  v8f acc[2][4];
#pragma unroll
  for (int f = 0; f < 2; ++f)
#pragma unroll
    for (int t = 0; t < 4; ++t) acc[f][t] = zero8();

  const _Float16* a0 = W1 + (size_t)(m0 + lm) * KPAD;
  const _Float16* a1 = W1 + (size_t)(m0 + 16 + lm) * KPAD;
  const _Float16* bb[4];
#pragma unroll
  for (int t = 0; t < 4; ++t)
    bb[t] = Xc + (size_t)(n0 + t * 16 + lm) * KPAD + koffB;

#pragma unroll
  for (int k0 = 0; k0 < KPAD; k0 += 32) {
    v16h af[2];
    af[0] = load_afrag(a0, k0, koffA);
    af[1] = load_afrag(a1, k0, koffA);
    v16h bf[4];
#pragma unroll
    for (int t = 0; t < 4; ++t) bf[t] = *(const v16h*)(bb[t] + k0);
#pragma unroll
    for (int f = 0; f < 2; ++f)
#pragma unroll
      for (int t = 0; t < 4; ++t) acc[f][t] = wmma_f16(af[f], bf[t], acc[f][t]);
  }

#pragma unroll
  for (int f = 0; f < 2; ++f)
#pragma unroll
    for (int t = 0; t < 4; ++t) {
      int n = n0 + t * 16 + lm;
      int mb = m0 + f * 16 + hi * 8;
      h8 pk;
#pragma unroll
      for (int r = 0; r < 8; ++r) {
        float v = acc[f][t][r] + bias[mb + r];
        pk[r] = (_Float16)(v > 0.0f ? v : 0.0f);
      }
      *(h8*)(Hout + (size_t)n * C1 + mb) = pk;  // NHWC, 16B store
    }
}

// ---------------- primary-caps conv (dominant: 196 GFLOP implicit GEMM) -----
// C[256 x 18432] = sum over 81 taps of Wtap[256x256] @ Htap; Hin is NHWC so the
// K (input-channel) dim is unit-stride: every B frag is one 32B load.
// All operands are L2-resident on MI455X (105MB + 10.6MB < 192MB L2).
__global__ __launch_bounds__(256) void prim_gemm_kernel(
    const _Float16* __restrict__ PT,   // [81][256][256]
    const _Float16* __restrict__ Hin,  // [(b*400+s)*256 + c]
    float* __restrict__ V) {           // [(b*36+s)*256 + m]
  const int lane = threadIdx.x & 31;
  const int wv = threadIdx.x >> 5;
  const int wm = wv >> 2, wn = wv & 3;
  const int m0 = blockIdx.x * 64 + wm * 32;
  const int n0 = blockIdx.y * 256 + wn * 64;
  const int lm = lane & 15;
  const int hi = (lane >> 4) & 1;
  const int koffA = hi * 8, koffB = hi * 16;

  int row0[4];  // NHWC spatial row index of output column's top-left input pixel
#pragma unroll
  for (int t = 0; t < 4; ++t) {
    int n = n0 + t * 16 + lm;
    int b = n / PPIX, s = n - b * PPIX;
    int py = s / HP, px = s - py * HP;
    row0[t] = b * HPIX + (2 * py) * HC + 2 * px;
  }

  v8f acc[2][4];
#pragma unroll
  for (int f = 0; f < 2; ++f)
#pragma unroll
    for (int t = 0; t < 4; ++t) acc[f][t] = zero8();

  const _Float16* arow0 = PT + (size_t)(m0 + lm) * C1;
  const _Float16* arow1 = PT + (size_t)(m0 + 16 + lm) * C1;

  for (int tap = 0; tap < KK; ++tap) {
    int ky = tap / 9, kx = tap - ky * 9;
    const _Float16* a0 = arow0 + (size_t)tap * (CPRIM * C1);
    const _Float16* a1 = arow1 + (size_t)tap * (CPRIM * C1);
    const _Float16* bb[4];
#pragma unroll
    for (int t = 0; t < 4; ++t)
      bb[t] = Hin + (size_t)(row0[t] + ky * HC + kx) * C1 + koffB;

#pragma unroll 4
    for (int kc = 0; kc < C1; kc += 32) {
      v16h af[2];
      af[0] = load_afrag(a0, kc, koffA);
      af[1] = load_afrag(a1, kc, koffA);
      v16h bf[4];
#pragma unroll
      for (int t = 0; t < 4; ++t) bf[t] = *(const v16h*)(bb[t] + kc);
#pragma unroll
      for (int f = 0; f < 2; ++f)
#pragma unroll
        for (int t = 0; t < 4; ++t) acc[f][t] = wmma_f16(af[f], bf[t], acc[f][t]);
    }
  }

  // channels contiguous -> whole accumulator fragment = one 32B store
#pragma unroll
  for (int f = 0; f < 2; ++f)
#pragma unroll
    for (int t = 0; t < 4; ++t) {
      int n = n0 + t * 16 + lm;
      *(v8f*)(V + (size_t)n * CPRIM + m0 + f * 16 + hi * 8) = acc[f][t];
    }
}

// ---------------- decoder MLP GEMM: out = act(A[MxK] @ W + bias) ------------
// Wt is pre-transposed [N][K] so B frags are K-contiguous 32B loads.
__global__ __launch_bounds__(256) void mlp_gemm_kernel(
    const _Float16* __restrict__ A, const _Float16* __restrict__ Wt,
    const float* __restrict__ bias, _Float16* __restrict__ out16,
    float* __restrict__ out32, int M, int N, int K, int act) {
  const int lane = threadIdx.x & 31;
  const int wv = threadIdx.x >> 5;
  const int wm = wv >> 2, wn = wv & 3;
  const int m0 = blockIdx.x * 64 + wm * 32;
  const int n0 = blockIdx.y * 256 + wn * 64;
  const int lm = lane & 15;
  const int hi = (lane >> 4) & 1;
  const int koffA = hi * 8, koffB = hi * 16;

  v8f acc[2][4];
#pragma unroll
  for (int f = 0; f < 2; ++f)
#pragma unroll
    for (int t = 0; t < 4; ++t) acc[f][t] = zero8();

  const _Float16* a0 = A + (size_t)(m0 + lm) * K;
  const _Float16* a1 = A + (size_t)(m0 + 16 + lm) * K;
  const _Float16* bb[4];
  bool bok[4];
#pragma unroll
  for (int t = 0; t < 4; ++t) {
    int n = n0 + t * 16 + lm;
    bok[t] = n < N;
    bb[t] = Wt + (size_t)(bok[t] ? n : 0) * K + koffB;
  }

#pragma unroll 4
  for (int k0 = 0; k0 < K; k0 += 32) {
    v16h af[2];
    af[0] = load_afrag(a0, k0, koffA);
    af[1] = load_afrag(a1, k0, koffA);
    v16h bf[4];
#pragma unroll
    for (int t = 0; t < 4; ++t) bf[t] = bok[t] ? *(const v16h*)(bb[t] + k0) : zero16h();
#pragma unroll
    for (int f = 0; f < 2; ++f)
#pragma unroll
      for (int t = 0; t < 4; ++t) acc[f][t] = wmma_f16(af[f], bf[t], acc[f][t]);
  }

#pragma unroll
  for (int f = 0; f < 2; ++f)
#pragma unroll
    for (int r = 0; r < 8; ++r) {
      int m = m0 + f * 16 + r + hi * 8;
#pragma unroll
      for (int t = 0; t < 4; ++t) {
        int n = n0 + t * 16 + lm;
        if (n < N) {
          float v = acc[f][t][r] + bias[n];
          if (act == 0) v = v > 0.0f ? v : 0.0f;       // relu
          else          v = 1.0f / (1.0f + expf(-v));  // sigmoid
          if (out16) out16[(size_t)m * N + n] = (_Float16)v;
          if (out32) out32[(size_t)m * N + n] = v;
        }
      }
    }
}

// ---------------- primary-caps routing (1 iter): act = squash(v/32 + b) -----
// V layout [(b*36+s)*256 + m]: the 8 atoms of a capsule are contiguous.
__global__ void prim_squash_kernel(const float* __restrict__ V,
                                   const float* __restrict__ pb,
                                   float* __restrict__ caps) {
  int idx = blockIdx.x * blockDim.x + threadIdx.x;  // B*32*36
  if (idx >= Bsz * NPRIM * PPIX) return;
  int b = idx / (NPRIM * PPIX);
  int r = idx - b * (NPRIM * PPIX);
  int o = r / PPIX;
  int s = r - o * PPIX;
  const float* vp = V + (size_t)(b * PPIX + s) * CPRIM + o * PAT;
  float pre[PAT];
  float n2 = 0.0f;
#pragma unroll
  for (int a = 0; a < PAT; ++a) {
    float v = vp[a] * (1.0f / 32.0f) + pb[o * PAT + a];
    pre[a] = v;
    n2 += v * v;
  }
  float sc = n2 / (1.0f + n2) * rsqrtf(n2 + 1e-8f);
  float* cp = caps + (size_t)(b * NCAPS + o * PPIX + s) * PAT;
#pragma unroll
  for (int a = 0; a < PAT; ++a) cp[a] = pre[a] * sc;
}

// ---------------- digit routing scalar kernels ------------------------------
__global__ void softmax10_kernel(const float* __restrict__ lg, float* __restrict__ rt) {
  int idx = blockIdx.x * blockDim.x + threadIdx.x;  // B*1152
  if (idx >= Bsz * NCAPS) return;
  const float* l = lg + (size_t)idx * NCLASS;
  float m = l[0];
#pragma unroll
  for (int o = 1; o < NCLASS; ++o) m = fmaxf(m, l[o]);
  float e[NCLASS], s = 0.0f;
#pragma unroll
  for (int o = 0; o < NCLASS; ++o) { e[o] = expf(l[o] - m); s += e[o]; }
  float inv = 1.0f / s;
  float* r = rt + (size_t)idx * NCLASS;
#pragma unroll
  for (int o = 0; o < NCLASS; ++o) r[o] = e[o] * inv;
}

// preact[b,o,a] = digit_b[o,a] + sum_i route[b,i,o] * (sum_p caps[b,i,p]*W[i,p,o*16+a])
__global__ void preact_kernel(const float* __restrict__ route,
                              const float* __restrict__ caps,
                              const float* __restrict__ dw,
                              const float* __restrict__ db,
                              float* __restrict__ pre) {
  int idx = blockIdx.x * blockDim.x + threadIdx.x;  // B*160
  if (idx >= Bsz * D0) return;
  int b = idx / D0;
  int j = idx - b * D0;
  int o = j >> 4;
  float acc = db[j];
  const float* rp = route + (size_t)b * NCAPS * NCLASS + o;
  const float* cp = caps + (size_t)b * NCAPS * PAT;
  const float* wp = dw + j;
  for (int i = 0; i < NCAPS; ++i) {
    float v = 0.0f;
#pragma unroll
    for (int p = 0; p < PAT; ++p) v += cp[p] * wp[p * D0];
    acc += rp[0] * v;
    rp += NCLASS;
    cp += PAT;
    wp += PAT * D0;
  }
  pre[idx] = acc;
}

__global__ void dsquash_kernel(const float* __restrict__ pre, float* __restrict__ act) {
  int idx = blockIdx.x * blockDim.x + threadIdx.x;  // B*10
  if (idx >= Bsz * NCLASS) return;
  const float* p = pre + (size_t)idx * DAT;
  float n2 = 0.0f;
#pragma unroll
  for (int a = 0; a < DAT; ++a) n2 += p[a] * p[a];
  float sc = n2 / (1.0f + n2) * rsqrtf(n2 + 1e-8f);
  float* d = act + (size_t)idx * DAT;
#pragma unroll
  for (int a = 0; a < DAT; ++a) d[a] = p[a] * sc;
}

// logits[b,i,o] += sum_p caps[b,i,p] * sum_a W[i,p,o*16+a]*act[b,o,a]
__global__ void logits_upd_kernel(const float* __restrict__ caps,
                                  const float* __restrict__ dw,
                                  const float* __restrict__ act,
                                  float* __restrict__ logits) {
  int idx = blockIdx.x * blockDim.x + threadIdx.x;  // B*1152*10
  if (idx >= Bsz * NCAPS * NCLASS) return;
  int b = idx / (NCAPS * NCLASS);
  int r = idx - b * (NCAPS * NCLASS);
  int i = r / NCLASS;
  int o = r - i * NCLASS;
  const float* ap = act + (size_t)b * D0 + o * DAT;
  const float* cp = caps + (size_t)(b * NCAPS + i) * PAT;
  const float* wp = dw + (size_t)i * (PAT * D0) + o * DAT;
  float s = 0.0f;
#pragma unroll
  for (int p = 0; p < PAT; ++p) {
    float w = 0.0f;
#pragma unroll
    for (int a = 0; a < DAT; ++a) w += wp[p * D0 + a] * ap[a];
    s += cp[p] * w;
  }
  logits[idx] += s;
}

// flat[b, o*16+a] = digit[b,o,a] * y[b,o]  (f16 for decoder GEMM)
__global__ void flat_kernel(const float* __restrict__ digit, const float* __restrict__ y,
                            _Float16* __restrict__ flat) {
  int idx = blockIdx.x * blockDim.x + threadIdx.x;  // B*160
  if (idx >= Bsz * D0) return;
  int b = idx / D0;
  int o = (idx - b * D0) >> 4;
  flat[idx] = (_Float16)(digit[idx] * y[b * NCLASS + o]);
}

}  // namespace

extern "C" void kernel_launch(void* const* d_in, const int* in_sizes, int n_in,
                              void* d_out, int out_size, void* d_ws, size_t ws_size,
                              hipStream_t stream) {
  (void)in_sizes; (void)n_in; (void)out_size; (void)ws_size;
  const float* x   = (const float*)d_in[0];
  const float* y   = (const float*)d_in[1];
  const float* w1  = (const float*)d_in[2];
  const float* b1  = (const float*)d_in[3];
  const float* pw  = (const float*)d_in[4];
  const float* pb  = (const float*)d_in[5];
  const float* dw  = (const float*)d_in[6];
  const float* db  = (const float*)d_in[7];
  const float* dc1 = (const float*)d_in[8];
  const float* dbc1= (const float*)d_in[9];
  const float* dc2 = (const float*)d_in[10];
  const float* dbc2= (const float*)d_in[11];
  const float* dc3 = (const float*)d_in[12];
  const float* dbc3= (const float*)d_in[13];
  float* out   = (float*)d_out;  // digit [512*160] ++ recon [512*784]
  float* digit = out;
  float* recon = out + Bsz * D0;
  char* ws = (char*)d_ws;

  size_t off = 0;
  auto grab = [&](size_t elems, size_t esz) -> char* {
    char* p = ws + off;
    off += (elems * esz + 255) & ~(size_t)255;
    return p;
  };
  _Float16* w1f    = (_Float16*)grab((size_t)C1 * KPAD, 2);            // 48 KB
  _Float16* xcol   = (_Float16*)grab((size_t)N1 * KPAD, 2);            // 39.3 MB
  _Float16* ptw    = (_Float16*)grab((size_t)KK * CPRIM * C1, 2);      // 10.6 MB
  _Float16* h16    = (_Float16*)grab((size_t)Bsz * HPIX * C1, 2);      // 105 MB (NHWC)
  float*    primv  = (float*)   grab((size_t)Bsz * PPIX * CPRIM, 4);   // 18.9 MB
  float*    caps   = (float*)   grab((size_t)Bsz * NCAPS * PAT, 4);    // 18.9 MB
  float*    logits = (float*)   grab((size_t)Bsz * NCAPS * NCLASS, 4); // 23.6 MB
  float*    route  = (float*)   grab((size_t)Bsz * NCAPS * NCLASS, 4); // 23.6 MB
  float*    pre    = (float*)   grab((size_t)Bsz * D0, 4);
  _Float16* dw1t   = (_Float16*)grab((size_t)512 * D0, 2);             // [512][160]
  _Float16* dw2t   = (_Float16*)grab((size_t)1024 * 512, 2);           // [1024][512]
  _Float16* dw3t   = (_Float16*)grab((size_t)NPIXI * 1024, 2);         // [784][1024]
  _Float16* flat16 = (_Float16*)grab((size_t)Bsz * D0, 2);
  _Float16* r1f    = (_Float16*)grab((size_t)Bsz * 512, 2);
  _Float16* r2f    = (_Float16*)grab((size_t)Bsz * 1024, 2);

  const int T = 256;
  auto blocks = [](long long n) { return (unsigned)((n + 255) / 256); };

  // ---- pack / convert / transpose ----
  pack_w1_kernel<<<blocks(C1 * KPAD), T, 0, stream>>>(w1, w1f);
  im2col1_kernel<<<blocks((long long)N1 * KPAD), T, 0, stream>>>(x, xcol);
  pack_pw_kernel<<<blocks((long long)KK * CPRIM * C1), T, 0, stream>>>(pw, ptw);
  tr_f16_kernel<<<blocks(D0 * 512), T, 0, stream>>>(dc1, dw1t, D0, 512);
  tr_f16_kernel<<<blocks(512 * 1024), T, 0, stream>>>(dc2, dw2t, 512, 1024);
  tr_f16_kernel<<<blocks(1024 * NPIXI), T, 0, stream>>>(dc3, dw3t, 1024, NPIXI);

  // ---- conv1 (WMMA GEMM, fused bias+relu, NHWC f16 out) ----
  conv1_gemm_kernel<<<dim3(C1 / 64, N1 / 256), T, 0, stream>>>(w1f, xcol, b1, h16);

  // ---- primary-caps conv (dominant WMMA GEMM: 256 x 20736 x 18432) ----
  prim_gemm_kernel<<<dim3(CPRIM / 64, N2 / 256), T, 0, stream>>>(ptw, h16, primv);

  // ---- primary caps routing (1 iter => squash(v/32 + b)) ----
  prim_squash_kernel<<<blocks(Bsz * NPRIM * PPIX), T, 0, stream>>>(primv, pb, caps);

  // ---- digit routing: 3 iterations, votes recomputed on the fly ----
  zero_f32_kernel<<<blocks(Bsz * NCAPS * NCLASS), T, 0, stream>>>(logits, Bsz * NCAPS * NCLASS);
  for (int it = 0; it < 3; ++it) {
    softmax10_kernel<<<blocks(Bsz * NCAPS), T, 0, stream>>>(logits, route);
    preact_kernel<<<blocks(Bsz * D0), T, 0, stream>>>(route, caps, dw, db, pre);
    dsquash_kernel<<<blocks(Bsz * NCLASS), T, 0, stream>>>(pre, digit);
    if (it < 2)
      logits_upd_kernel<<<blocks(Bsz * NCAPS * NCLASS), T, 0, stream>>>(caps, dw, digit, logits);
  }

  // ---- decoder: mask, then 3 WMMA GEMMs ----
  flat_kernel<<<blocks(Bsz * D0), T, 0, stream>>>(digit, y, flat16);
  mlp_gemm_kernel<<<dim3(Bsz / 64, (512 + 255) / 256), T, 0, stream>>>(
      flat16, dw1t, dbc1, r1f, nullptr, Bsz, 512, D0, 0);
  mlp_gemm_kernel<<<dim3(Bsz / 64, (1024 + 255) / 256), T, 0, stream>>>(
      r1f, dw2t, dbc2, r2f, nullptr, Bsz, 1024, 512, 0);
  mlp_gemm_kernel<<<dim3(Bsz / 64, (NPIXI + 255) / 256), T, 0, stream>>>(
      r2f, dw3t, dbc3, nullptr, recon, Bsz, NPIXI, 1024, 1);
}
